// Attention_26869315404520
// MI455X (gfx1250) — matrix operand order repaired and checked by reading, not compiled, not run on hardware
//
#include <hip/hip_runtime.h>
#include <hip/hip_bf16.h>
#include <math.h>

// ---------------- problem constants ----------------
#define Ctot   192
#define Dq     24          // d_qk == d_v
#define Dpad   32          // padded to WMMA K-dim
#define Ltot   4096        // H*W
#define Btot   8
#define TOKS   (Btot*Ltot) // 32768

typedef __bf16 bf16_t;
typedef __bf16 v16bf __attribute__((ext_vector_type(16)));
typedef __bf16 v8bf  __attribute__((ext_vector_type(8)));
typedef float  v8f   __attribute__((ext_vector_type(8)));

#if __has_builtin(__builtin_amdgcn_global_load_async_to_lds_b64) && \
    __has_builtin(__builtin_amdgcn_s_wait_asynccnt)
#define USE_ASYNC_LDS 1
#else
#define USE_ASYNC_LDS 0
#endif

// exact builtin parameter types (from the round-2 diagnostic):
// (v2i __device__*, v2i __shared__*, imm int, imm int)
typedef int v2i __attribute__((ext_vector_type(2)));
typedef __attribute__((address_space(1))) v2i* g64p;
typedef __attribute__((address_space(3))) v2i* l64p;

// =====================================================================
// Kernel 1: QKV projection + bf16 pack.
//   Qb, Kb : [token][32] bf16   (d = 0..23 real, 24..31 zero)
//   Vt     : [b][32][L]  bf16   (transposed: dv-major, rows 24..31 zero)
// =====================================================================
#define K1_TOK 128

__global__ __launch_bounds__(256) void qkv_pack_kernel(
    const float* __restrict__ X,
    const float* __restrict__ wq, const float* __restrict__ wk,
    const float* __restrict__ wv,
    bf16_t* __restrict__ Qb, bf16_t* __restrict__ Kb, bf16_t* __restrict__ Vt)
{
    __shared__ __align__(16) float lds_w[3][Ctot*Dq];   // 3*18KB
    __shared__ __align__(16) float lds_x[K1_TOK][Ctot]; // 98KB
    const int tid  = threadIdx.x;
    const int tok0 = blockIdx.x * K1_TOK;

    for (int i = tid; i < Ctot*Dq; i += 256) {
        lds_w[0][i] = wq[i]; lds_w[1][i] = wk[i]; lds_w[2][i] = wv[i];
    }
    const float4* xsrc = (const float4*)(X + (size_t)tok0 * Ctot);
    float4* xdst = (float4*)&lds_x[0][0];
    for (int i = tid; i < (K1_TOK*Ctot)/4; i += 256) xdst[i] = xsrc[i];
    __syncthreads();

    for (int it = 0; it < (K1_TOK*Dq)/256; ++it) {     // 12 iters
        const int idx = it*256 + tid;
        const int t = idx / Dq;
        const int d = idx % Dq;
        float aq = 0.f, ak = 0.f, av = 0.f;
        const float* xr = &lds_x[t][0];
        #pragma unroll 4
        for (int c = 0; c < Ctot; ++c) {
            const float x = xr[c];
            aq += x * lds_w[0][c*Dq + d];
            ak += x * lds_w[1][c*Dq + d];
            av += x * lds_w[2][c*Dq + d];
        }
        const int tok = tok0 + t;
        const int b   = tok >> 12;       // / 4096
        const int l   = tok & (Ltot-1);
        Qb[(size_t)tok*Dpad + d] = (bf16_t)aq;
        Kb[(size_t)tok*Dpad + d] = (bf16_t)ak;
        Vt[((size_t)b*Dpad + d)*Ltot + l] = (bf16_t)av;
        if (d < (Dpad - Dq)) {           // d in 0..7 clears pad lanes 24..31
            Qb[(size_t)tok*Dpad + Dq + d] = (bf16_t)0.0f;
            Kb[(size_t)tok*Dpad + Dq + d] = (bf16_t)0.0f;
            Vt[((size_t)b*Dpad + Dq + d)*Ltot + l] = (bf16_t)0.0f;
        }
    }
}

// =====================================================================
// Kernel 2: flash attention with v_wmma_f32_16x16x32_bf16
//   grid = B * (L/BM); 256 threads = 8 waves; wave w owns 16 query rows
//   Double-buffered K/V staging; async global->LDS when available.
// =====================================================================
#define BM    128
#define BN    32
#define NWAVE 8
#define SP    33   // padded S-row stride (bank-conflict avoidance)
#define NBLK  (Ltot/BN)

__global__ __launch_bounds__(256) void flash_attn_kernel(
    const float* __restrict__ X,  const float* __restrict__ wo,
    const float* __restrict__ gamma_p,
    const bf16_t* __restrict__ Qb, const bf16_t* __restrict__ Kb,
    const bf16_t* __restrict__ Vt, float* __restrict__ Out)
{
    __shared__ __align__(16) float  lds_wo[Dq*Ctot];        // 18KB
    __shared__ __align__(64) bf16_t lds_k[2][BN*Dpad];      // 2x2KB [key][dq]
    __shared__ __align__(64) bf16_t lds_v[2][Dpad*BN];      // 2x2KB [dv][key]
    __shared__ __align__(16) float  lds_s[NWAVE][16*SP];    // ~17KB
    __shared__ __align__(64) bf16_t lds_p[NWAVE][16*BN];    // 8KB
    __shared__ __align__(16) float  lds_a[NWAVE][16];       // alpha / 1/l
    __shared__ __align__(16) float  lds_o[BM*Dq];           // 12KB

    const int tid  = threadIdx.x;
    const int w    = tid >> 5;
    const int lane = tid & 31;
    const int hi   = lane >> 4;      // 0: lanes 0-15, 1: lanes 16-31
    const int ln   = lane & 15;
    const int b    = blockIdx.x >> 5;       // 32 q-blocks per batch
    const int row0 = (blockIdx.x & 31) * BM;

    for (int i = tid; i < Dq*Ctot; i += 256) lds_wo[i] = wo[i];

    // A-operand (Q, 16x32 bf16): lane ln holds row ln,
    // contiguous K-chunk {hi*16 .. hi*16+15}
    const size_t qrow = (size_t)(b*Ltot + row0 + w*16 + ln) * Dpad;
    const v8bf q_lo = *(const v8bf*)(Qb + qrow + hi*16);
    const v8bf q_hi = *(const v8bf*)(Qb + qrow + hi*16 + 8);
    const v16bf aQ = __builtin_shufflevector(q_lo, q_hi,
        0,1,2,3,4,5,6,7,8,9,10,11,12,13,14,15);

    v8f o0 = {}; v8f o1 = {};
    float m_run = -INFINITY, l_run = 0.f;

    const bf16_t* Kbb = Kb + (size_t)b*Ltot*Dpad;
    const bf16_t* Vtb = Vt + (size_t)b*Dpad*Ltot;

    // per-thread staging addresses: 8 bytes (4 bf16) each, branchless
    const int vrow = tid >> 3;          // 0..31 (dv row)
    const int vcol = (tid & 7) * 4;     // 0,4,...,28 (key within block)

    // ---- stage block `kb` into buffer `bb` ----
    auto stage = [&](int kb, int bb) {
        const bf16_t* kg = Kbb + (size_t)kb*BN*Dpad + tid*4;
        const bf16_t* vg = Vtb + (size_t)vrow*Ltot + kb*BN + vcol;
        bf16_t* kl = &lds_k[bb][tid*4];
        bf16_t* vl = &lds_v[bb][vrow*BN + vcol];
#if USE_ASYNC_LDS
        __builtin_amdgcn_global_load_async_to_lds_b64((g64p)kg, (l64p)kl, 0, 0);
        __builtin_amdgcn_global_load_async_to_lds_b64((g64p)vg, (l64p)vl, 0, 0);
#else
        *(uint2*)kl = *(const uint2*)kg;
        *(uint2*)vl = *(const uint2*)vg;
#endif
        if (kb + 2 < NBLK) {            // gfx1250 global_prefetch_b8
            __builtin_prefetch(Kbb + (size_t)(kb+2)*BN*Dpad + tid*4, 0, 3);
            __builtin_prefetch(Vtb + (size_t)vrow*Ltot + (kb+2)*BN, 0, 3);
        }
    };
    auto stage_done = [&]() {
#if USE_ASYNC_LDS
        __builtin_amdgcn_s_wait_asynccnt(0);
#endif
        __syncthreads();
    };

    stage(0, 0);
    stage_done();

    for (int kb = 0; kb < NBLK; ++kb) {
        const int cur = kb & 1;
        if (kb + 1 < NBLK) stage(kb + 1, cur ^ 1);

        // ---- S = Q * K^T : two 16x16 tiles over keys ----
        // B-operand: lane ln = column (key), contiguous K-chunk hi*16..+15
        const v16bf bK0 = *(const v16bf*)(&lds_k[cur][(ln     )*Dpad + hi*16]);
        const v16bf bK1 = *(const v16bf*)(&lds_k[cur][(16 + ln)*Dpad + hi*16]);
        const v8f zc = {};
        v8f s0 = __builtin_amdgcn_wmma_f32_16x16x32_bf16(
                     false, aQ, false, bK0, (short)0, zc, false, false);
        v8f s1 = __builtin_amdgcn_wmma_f32_16x16x32_bf16(
                     false, aQ, false, bK1, (short)0, zc, false, false);

        // ---- spill S to per-wave LDS tile (C-layout: VGPR r -> row r+8*hi) ----
        float* sw = &lds_s[w][0];
        #pragma unroll
        for (int r = 0; r < 8; ++r) {
            const int row = r + hi*8;
            sw[row*SP + ln]      = s0[r];
            sw[row*SP + 16 + ln] = s1[r];
        }
        __builtin_amdgcn_wave_barrier();

        // ---- online softmax: lanes 0..15, one row each ----
        if (lane < 16) {
            float sv[BN];
            float mloc = -INFINITY;
            #pragma unroll
            for (int j = 0; j < BN; ++j) {
                sv[j] = sw[lane*SP + j];
                mloc = fmaxf(mloc, sv[j]);
            }
            const float m_new = fmaxf(m_run, mloc);
            const float alpha = __expf(m_run - m_new);
            float lsum = 0.f;
            #pragma unroll
            for (int j = 0; j < BN; ++j) {
                const float p = __expf(sv[j] - m_new);
                lsum += p;
                lds_p[w][lane*BN + j] = (bf16_t)p;
            }
            l_run = l_run * alpha + lsum;
            m_run = m_new;
            lds_a[w][lane] = alpha;
        }
        __builtin_amdgcn_wave_barrier();

        // ---- rescale accumulators by alpha(row) ----
        #pragma unroll
        for (int r = 0; r < 8; ++r) {
            const float a = lds_a[w][r + hi*8];
            o0[r] *= a;  o1[r] *= a;
        }

        // ---- O += P * V : A = P(16x32), B = Vt tiles (dv 0..15 / 16..31) ----
        const v8bf p_lo = *(const v8bf*)(&lds_p[w][ln*BN + hi*16]);
        const v8bf p_hi = *(const v8bf*)(&lds_p[w][ln*BN + hi*16 + 8]);
        const v16bf aP = __builtin_shufflevector(p_lo, p_hi,
            0,1,2,3,4,5,6,7,8,9,10,11,12,13,14,15);
        const v16bf bV0 = *(const v16bf*)(&lds_v[cur][(ln     )*BN + hi*16]);
        const v16bf bV1 = *(const v16bf*)(&lds_v[cur][(16 + ln)*BN + hi*16]);
        o0 = __builtin_amdgcn_wmma_f32_16x16x32_bf16(
                 false, aP, false, bV0, (short)0, o0, false, false);
        o1 = __builtin_amdgcn_wmma_f32_16x16x32_bf16(
                 false, aP, false, bV1, (short)0, o1, false, false);

        stage_done();   // async writes of next buffer done + all waves done reading cur
    }

    // ---- finalize: O /= l ; spill rows to lds_o ----
    if (lane < 16) lds_a[w][lane] = (l_run > 0.f) ? (1.f / l_run) : 0.f;
    __builtin_amdgcn_wave_barrier();
    #pragma unroll
    for (int r = 0; r < 8; ++r) {
        const float iv = lds_a[w][r + hi*8];
        const int row = w*16 + r + hi*8;
        lds_o[row*Dq + ln] = o0[r] * iv;
        if (ln < 8) lds_o[row*Dq + 16 + ln] = o1[r] * iv;
    }
    __syncthreads();

    // ---- epilogue: out = X + gamma * (O @ Wo) ----
    const float gamma = *gamma_p;
    const size_t obase = (size_t)(b*Ltot + row0) * Ctot;
    for (int i = tid; i < BM*Ctot; i += 256) {
        const int row = i / Ctot, c = i - row*Ctot;
        float acc = 0.f;
        #pragma unroll
        for (int d = 0; d < Dq; ++d)
            acc += lds_o[row*Dq + d] * lds_wo[d*Ctot + c];
        Out[obase + i] = X[obase + i] + gamma * acc;
    }
}

// =====================================================================
extern "C" void kernel_launch(void* const* d_in, const int* in_sizes, int n_in,
                              void* d_out, int out_size, void* d_ws, size_t ws_size,
                              hipStream_t stream)
{
    const float* X  = (const float*)d_in[0];
    const float* wq = (const float*)d_in[1];
    const float* wk = (const float*)d_in[2];
    const float* wv = (const float*)d_in[3];
    const float* wo = (const float*)d_in[4];
    const float* gp = (const float*)d_in[5];
    float* Out = (float*)d_out;

    char* ws = (char*)d_ws;
    const size_t seg = (size_t)TOKS * Dpad * sizeof(bf16_t);  // 2 MB
    bf16_t* Qb = (bf16_t*)(ws);
    bf16_t* Kb = (bf16_t*)(ws + seg);
    bf16_t* Vt = (bf16_t*)(ws + 2*seg);

    qkv_pack_kernel<<<TOKS / K1_TOK, 256, 0, stream>>>(X, wq, wk, wv, Qb, Kb, Vt);
    flash_attn_kernel<<<Btot * (Ltot / BM), 256, 0, stream>>>(X, wo, gp, Qb, Kb, Vt, Out);
}
